// Net_Att_35072702939688
// MI455X (gfx1250) — matrix operand order repaired and checked by reading, not compiled, not run on hardware
//
#include <hip/hip_runtime.h>
#include <hip/hip_bf16.h>

// ---------------------------------------------------------------------------
// Types for CDNA5 WMMA (wave32, 16x16x32 bf16 -> f32)
// ---------------------------------------------------------------------------
typedef __attribute__((ext_vector_type(16))) __bf16 v16bf;
typedef __attribute__((ext_vector_type(8)))  float  v8f;
typedef int int4v __attribute__((vector_size(16)));   // matches async-LDS builtin params

union Frag { uint4 u[2]; v16bf v; };

#define T_TOK 196608   // 32768 * 6 tokens
#define NBATCH 32768
#define NCH 256

// Async global->LDS staging path (gfx1250 ASYNCcnt-tracked DMA) when available.
#if defined(__gfx1250__) && __has_builtin(__builtin_amdgcn_global_load_async_to_lds_b128)
#define USE_ASYNC_LDS 1
#else
#define USE_ASYNC_LDS 0
#endif

__device__ __forceinline__ unsigned short f2b(float f) {
  unsigned int u = __float_as_uint(f);
  unsigned int r = (u + 0x7FFFu + ((u >> 16) & 1u)) >> 16;  // RNE
  return (unsigned short)r;
}
__device__ __forceinline__ float b2f(unsigned short h) {
  return __uint_as_float(((unsigned int)h) << 16);
}

// ---------------------------------------------------------------------------
// Packing / conversion kernels
// ---------------------------------------------------------------------------
// x (B,6,6,3,3) f32 -> Xpad bf16 [T][64]  (54 real channels + 10 zero pad)
__global__ void pack_x_kernel(const float* __restrict__ x,
                              unsigned short* __restrict__ xp) {
  int idx = blockIdx.x * 256 + threadIdx.x;   // t*64 + c
  int t = idx >> 6, c = idx & 63;
  float v = 0.f;
  if (c < 54) {
    int b = t / 6, p = t % 6;
    int i = c / 9, jk = c % 9, j = jk / 3, k = jk % 3;
    v = x[((((size_t)b * 6 + i) * 6 + p) * 3 + j) * 3 + k];
  }
  xp[idx] = f2b(v);
}

__global__ void f32_to_bf16_kernel(const float* __restrict__ s,
                                   unsigned short* __restrict__ d, int n) {
  int i = blockIdx.x * 256 + threadIdx.x;
  if (i < n) d[i] = f2b(s[i]);
}

// ec_w (256x54) -> bf16 [256][64] zero padded
__global__ void pack_ecw_kernel(const float* __restrict__ w,
                                unsigned short* __restrict__ d) {
  int i = blockIdx.x * 256 + threadIdx.x;     // 256*64
  int o = i >> 6, c = i & 63;
  d[i] = f2b(c < 54 ? w[o * 54 + c] : 0.f);
}

// f1_w [512][1536] with inner index c*6+p  ->  Wf1p [512][1536] inner p*256+c
__global__ void pack_f1w_kernel(const float* __restrict__ w,
                                unsigned short* __restrict__ d, int n) {
  int i = blockIdx.x * 256 + threadIdx.x;
  if (i >= n) return;
  int o = i / 1536, r = i % 1536, p = r >> 8, c = r & 255;
  d[i] = f2b(w[o * 1536 + c * 6 + p]);
}

__global__ void zero_kernel(float* p, int n) {
  int i = blockIdx.x * 256 + threadIdx.x;
  if (i < n) p[i] = 0.f;
}

// ---------------------------------------------------------------------------
// BatchNorm (training-mode, reduction over all T tokens per channel)
// ---------------------------------------------------------------------------
__global__ __launch_bounds__(256) void bn_stats_kernel(
    const float* __restrict__ H, float* __restrict__ stats, int T) {
  int c = threadIdx.x;                           // 256 channels
  float s = 0.f, s2 = 0.f;
  for (int t = blockIdx.x; t < T; t += gridDim.x) {
    float v = H[(size_t)t * NCH + c];
    s += v; s2 += v * v;
  }
  atomicAdd(&stats[c], s);
  atomicAdd(&stats[NCH + c], s2);
}

__global__ void bn_finalize_kernel(const float* __restrict__ stats,
                                   const float* __restrict__ g,
                                   const float* __restrict__ b,
                                   float* __restrict__ prm, float invT) {
  int c = threadIdx.x;
  float m   = stats[c] * invT;
  float var = stats[NCH + c] * invT - m * m;
  float sc  = g[c] * rsqrtf(var + 1e-5f);
  prm[c]        = sc;
  prm[NCH + c]  = b[c] - m * sc;
}

template <int RELU>
__global__ __launch_bounds__(256) void bn_apply_kernel(
    const float* __restrict__ H, const float* __restrict__ prm,
    unsigned short* __restrict__ Hb, int total) {
  int i = blockIdx.x * 256 + threadIdx.x;
  if (i >= total) return;
  int c = i & 255;
  float v = prm[c] * H[i] + prm[NCH + c];
  if (RELU) v = v > 0.f ? v : 0.f;
  Hb[i] = f2b(v);
}

// ---------------------------------------------------------------------------
// Generic NT GEMM on WMMA:  out[n][m] (+)= sum_k W[m][k] * X[n][k]  (bf16 in)
// Block = 256 threads = 8 waves. Wave tile: 64(M) x 32(N) = 4x2 fragments.
// All 8 waves share the same 64-row W tile -> stage it once per block in LDS
// via async global->LDS DMA (ASYNCcnt), A-fragments then come from
// ds_load_b128 instead of 8x-redundant global traffic.
// Grid: (N/256, M/64).  K % 32 == 0, M % 64 == 0, N % 256 == 0.
// EPI: 0 = bias + embed_p, f32 out   (embed)
//      1 = bias, f32 out             (qkv)
//      2 = bias + relu, bf16 out     (d1, f1)
//      3 = bias + accumulate f32 out (d2 residual)
// ---------------------------------------------------------------------------
#define KC 128          // K-chunk staged per barrier (halves)
#define LDSW 136        // LDS row stride in halves (+8 pad -> 272B, kills bank conflicts)

template <int EPI>
__global__ __launch_bounds__(256) void gemm_nt_kernel(
    const unsigned short* __restrict__ W,  // [M][K] bf16
    const unsigned short* __restrict__ X,  // [N][K] bf16
    const float* __restrict__ bias,        // [M]
    const float* __restrict__ embed,       // [M*6] (EPI==0) else unused
    float* __restrict__ outF,              // [N][ldD] f32
    unsigned short* __restrict__ outB,     // [N][ldD] bf16
    int K, int ldD) {
  __shared__ __align__(16) unsigned short sW[64 * LDSW];  // 17408 B

  int tid  = threadIdx.x;
  int lane = tid & 31;
  int wave = tid >> 5;
  int n0 = blockIdx.x * 256 + wave * 32;
  int m0 = blockIdx.y * 64;
  int lhi = lane >> 4;       // 0/1 half-wave
  int llo = lane & 15;

  // B fragment (32x16 bf16): lane holds 16 contiguous halves of X row
  // (n0+nf*16+llo) starting at k = lhi*16.
  const unsigned short* xcol[2];
#pragma unroll
  for (int nf = 0; nf < 2; ++nf)
    xcol[nf] = X + (size_t)(n0 + nf * 16 + llo) * K + lhi * 16;

  v8f acc[4][2];
#pragma unroll
  for (int mf = 0; mf < 4; ++mf)
#pragma unroll
    for (int nf = 0; nf < 2; ++nf) acc[mf][nf] = (v8f)(0.f);

  for (int kc = 0; kc < K; kc += KC) {
    const int kcur = (K - kc) < KC ? (K - kc) : KC;   // 64 or 128 here
    // ---- stage W[m0..m0+63][kc..kc+kcur) into LDS (16B per thread-iter) ----
    const int nchunks = (64 * kcur) >> 3;             // # of 8-half chunks
    const int cpr = kcur >> 3;                        // chunks per row
    for (int i = tid; i < nchunks; i += 256) {
      int row = i / cpr;
      int c8  = (i - row * cpr) << 3;
      const unsigned short* g = W + (size_t)(m0 + row) * K + kc + c8;
      unsigned short* l = &sW[row * LDSW + c8];
#if USE_ASYNC_LDS
      __builtin_amdgcn_global_load_async_to_lds_b128(
          (__attribute__((address_space(1))) int4v*)g,
          (__attribute__((address_space(3))) int4v*)l, 0, 0);
#else
      *reinterpret_cast<uint4*>(l) = *reinterpret_cast<const uint4*>(g);
#endif
    }
#if USE_ASYNC_LDS
#if __has_builtin(__builtin_amdgcn_s_wait_asynccnt)
    __builtin_amdgcn_s_wait_asynccnt(0);
#else
    asm volatile("s_wait_asynccnt 0" ::: "memory");
#endif
#endif
    __syncthreads();

    // ---- compute on the staged chunk ----
    for (int kk = 0; kk < kcur; kk += 32) {
      Frag bf[2];
#pragma unroll
      for (int nf = 0; nf < 2; ++nf) {
        const unsigned short* p = xcol[nf] + kc + kk;
        bf[nf].u[0] = *reinterpret_cast<const uint4*>(p);
        bf[nf].u[1] = *reinterpret_cast<const uint4*>(p + 8);
        __builtin_prefetch((const void*)(p + 32), 0, 1);  // global_prefetch_b8
      }
#pragma unroll
      for (int mf = 0; mf < 4; ++mf) {
        // A fragment (16x32): lane holds row m = mf*16+llo, 16 contiguous halves
        // at k = kk + lhi*16.  Served from LDS (ds_load_b128).
        Frag af;
        const unsigned short* l = &sW[(mf * 16 + llo) * LDSW + kk + lhi * 16];
        af.u[0] = *reinterpret_cast<const uint4*>(l);
        af.u[1] = *reinterpret_cast<const uint4*>(l + 8);
#pragma unroll
        for (int nf = 0; nf < 2; ++nf) {
          acc[mf][nf] = __builtin_amdgcn_wmma_f32_16x16x32_bf16(
              false, af.v, false, bf[nf].v, (short)0, acc[mf][nf], false, false);
        }
      }
    }
    __syncthreads();
  }

  // Epilogue. C/D layout: lane holds col n = llo, rows m = mf*16 + lhi*8 + r.
#pragma unroll
  for (int mf = 0; mf < 4; ++mf) {
#pragma unroll
    for (int nf = 0; nf < 2; ++nf) {
      int n  = n0 + nf * 16 + llo;
      int mb = m0 + mf * 16 + (lhi << 3);
      v8f a = acc[mf][nf];
      float vals[8];
#pragma unroll
      for (int r = 0; r < 8; ++r) {
        float v = a[r] + bias[mb + r];
        if (EPI == 0) v += embed[(mb + r) * 6 + (n % 6)];
        if (EPI == 2) v = v > 0.f ? v : 0.f;
        vals[r] = v;
      }
      if (EPI == 2) {
        uint4 o;
        o.x = (unsigned)f2b(vals[0]) | ((unsigned)f2b(vals[1]) << 16);
        o.y = (unsigned)f2b(vals[2]) | ((unsigned)f2b(vals[3]) << 16);
        o.z = (unsigned)f2b(vals[4]) | ((unsigned)f2b(vals[5]) << 16);
        o.w = (unsigned)f2b(vals[6]) | ((unsigned)f2b(vals[7]) << 16);
        *reinterpret_cast<uint4*>(outB + (size_t)n * ldD + mb) = o;
      } else if (EPI == 3) {
        float* p = outF + (size_t)n * ldD + mb;
        float4 lo = *reinterpret_cast<float4*>(p);
        float4 hi = *reinterpret_cast<float4*>(p + 4);
        lo.x += vals[0]; lo.y += vals[1]; lo.z += vals[2]; lo.w += vals[3];
        hi.x += vals[4]; hi.y += vals[5]; hi.z += vals[6]; hi.w += vals[7];
        *reinterpret_cast<float4*>(p)     = lo;
        *reinterpret_cast<float4*>(p + 4) = hi;
      } else {
        float* p = outF + (size_t)n * ldD + mb;
        *reinterpret_cast<float4*>(p)     = make_float4(vals[0], vals[1], vals[2], vals[3]);
        *reinterpret_cast<float4*>(p + 4) = make_float4(vals[4], vals[5], vals[6], vals[7]);
      }
    }
  }
}

// ---------------------------------------------------------------------------
// Attention: per sample (6 tokens), q,k 32x6, v 256x6, softmax over axis i.
// 64 threads / sample, 4 samples / block. Fused residual add into H.
// QKV layout: [t][320] f32.  H layout: [t][256] f32.
// ---------------------------------------------------------------------------
__global__ __launch_bounds__(256) void attention_kernel(
    const float* __restrict__ QKV, float* __restrict__ H) {
  __shared__ float sq[4][32][6];
  __shared__ float sk[4][32][6];
  __shared__ float sb[4][6][6];    // beta[i][j]
  int ls  = threadIdx.x & 63;
  int smp = threadIdx.x >> 6;
  int b   = blockIdx.x * 4 + smp;
  const float* qkv = QKV + (size_t)b * 6 * 320;

  for (int i = ls; i < 384; i += 64) {     // 2 * 32 * 6 values
    int c = i & 31, rest = i >> 5, p = rest % 6, which = rest / 6;
    float v = qkv[(size_t)p * 320 + which * 32 + c];
    if (which == 0) sq[smp][c][p] = v; else sk[smp][c][p] = v;
  }
  __syncthreads();

  if (ls < 36) {
    int i = ls / 6, j = ls % 6;
    float s = 0.f;
    for (int c = 0; c < 32; ++c) s += sq[smp][c][i] * sk[smp][c][j];
    sb[smp][i][j] = s * 0.17677669529663687f;   // 1/sqrt(32)
  }
  __syncthreads();

  if (ls < 6) {                 // softmax over i for column j = ls
    int j = ls;
    float mx = sb[smp][0][j];
    for (int i = 1; i < 6; ++i) mx = fmaxf(mx, sb[smp][i][j]);
    float sum = 0.f, e[6];
    for (int i = 0; i < 6; ++i) { e[i] = __expf(sb[smp][i][j] - mx); sum += e[i]; }
    float inv = 1.f / sum;
    for (int i = 0; i < 6; ++i) sb[smp][i][j] = e[i] * inv;
  }
  __syncthreads();

  for (int o = ls; o < 1536; o += 64) {    // a[c][j], c = o%256, j = o/256
    int c = o & 255, j = o >> 8;
    float s = 0.f;
    for (int i = 0; i < 6; ++i) s += qkv[(size_t)i * 320 + 64 + c] * sb[smp][i][j];
    H[((size_t)b * 6 + j) * NCH + c] += s;
  }
}

// ---------------------------------------------------------------------------
// Final head: out[b] = 10 * (dot(G1[b][0:512], f2_w) + f2_b)
// ---------------------------------------------------------------------------
__global__ __launch_bounds__(256) void f2_final_kernel(
    const unsigned short* __restrict__ G1, const float* __restrict__ f2w,
    const float* __restrict__ f2b_, float* __restrict__ out) {
  int b = blockIdx.x * 256 + threadIdx.x;
  const unsigned short* g = G1 + (size_t)b * 512;
  float s = 0.f;
#pragma unroll 8
  for (int i = 0; i < 512; ++i) s += b2f(g[i]) * f2w[i];
  out[b] = 10.f * (s + f2b_[0]);
}

// ---------------------------------------------------------------------------
// Host orchestration
// ---------------------------------------------------------------------------
extern "C" void kernel_launch(void* const* d_in, const int* in_sizes, int n_in,
                              void* d_out, int out_size, void* d_ws, size_t ws_size,
                              hipStream_t stream) {
  (void)in_sizes; (void)n_in; (void)out_size; (void)ws_size;
  const float* x       = (const float*)d_in[0];
  const float* ec_w    = (const float*)d_in[1];
  const float* ec_b    = (const float*)d_in[2];
  const float* embed_p = (const float*)d_in[3];
  const float* f1_w = (const float*)d_in[24];
  const float* f1_b = (const float*)d_in[25];
  const float* f2_w = (const float*)d_in[26];
  const float* f2_b = (const float*)d_in[27];

  char* ws = (char*)d_ws;
  size_t off = 0;
  auto carve = [&](size_t bytes) -> char* {
    char* p = ws + off;
    off = (off + bytes + 255) & ~(size_t)255;
    return p;
  };
  const size_t T = T_TOK;
  float*          H    = (float*)         carve(T * NCH * 4);   // fp32 residual
  unsigned short* Hb   = (unsigned short*)carve(T * NCH * 2);   // bf16 operand
  float*          QKV  = (float*)         carve(T * 320 * 4);
  unsigned short* Y1   = (unsigned short*)QKV;                  // alias (d1 out)
  unsigned short* G1   = (unsigned short*)carve((size_t)NBATCH * 512 * 2);
  unsigned short* Xpad = (unsigned short*)carve(T * 64 * 2);
  unsigned short* Wec  = (unsigned short*)carve(256 * 64 * 2);
  unsigned short* Wq[2]  = {(unsigned short*)carve(320 * 256 * 2),
                            (unsigned short*)carve(320 * 256 * 2)};
  unsigned short* Wd1[2] = {(unsigned short*)carve(256 * 256 * 2),
                            (unsigned short*)carve(256 * 256 * 2)};
  unsigned short* Wd2[2] = {(unsigned short*)carve(256 * 256 * 2),
                            (unsigned short*)carve(256 * 256 * 2)};
  unsigned short* Wf1p = (unsigned short*)carve(512 * 1536 * 2);
  float* stats = (float*)carve(512 * 4);
  float* prm   = (float*)carve(512 * 4);

  // ---- weight / input staging -------------------------------------------
  pack_x_kernel<<<(T * 64) / 256, 256, 0, stream>>>(x, Xpad);
  pack_ecw_kernel<<<(256 * 64) / 256, 256, 0, stream>>>(ec_w, Wec);
  for (int L = 0; L < 2; ++L) {
    const float* qkv_w = (const float*)d_in[4 + 10 * L + 0];
    const float* d1_w  = (const float*)d_in[4 + 10 * L + 6];
    const float* d2_w  = (const float*)d_in[4 + 10 * L + 8];
    f32_to_bf16_kernel<<<(320 * 256 + 255) / 256, 256, 0, stream>>>(qkv_w, Wq[L], 320 * 256);
    f32_to_bf16_kernel<<<(256 * 256 + 255) / 256, 256, 0, stream>>>(d1_w, Wd1[L], 256 * 256);
    f32_to_bf16_kernel<<<(256 * 256 + 255) / 256, 256, 0, stream>>>(d2_w, Wd2[L], 256 * 256);
  }
  pack_f1w_kernel<<<(512 * 1536 + 255) / 256, 256, 0, stream>>>(f1_w, Wf1p, 512 * 1536);

  const int NBLK = (int)(T / 256);       // 768 n-blocks over tokens
  const float invT = 1.0f / (float)T;
  const int TOT = (int)(T * NCH);

  // ---- embed: H = ec_w @ x + ec_b + embed_p -----------------------------
  gemm_nt_kernel<0><<<dim3(NBLK, 256 / 64), 256, 0, stream>>>(
      Wec, Xpad, ec_b, embed_p, H, nullptr, 64, NCH);

  // ---- two attention layers ---------------------------------------------
  for (int L = 0; L < 2; ++L) {
    const float* qkv_b = (const float*)d_in[4 + 10 * L + 1];
    const float* bn1_g = (const float*)d_in[4 + 10 * L + 2];
    const float* bn1_b = (const float*)d_in[4 + 10 * L + 3];
    const float* bn2_g = (const float*)d_in[4 + 10 * L + 4];
    const float* bn2_b = (const float*)d_in[4 + 10 * L + 5];
    const float* d1_b  = (const float*)d_in[4 + 10 * L + 7];
    const float* d2_b  = (const float*)d_in[4 + 10 * L + 9];

    // bn1 + relu -> Hb
    zero_kernel<<<2, 256, 0, stream>>>(stats, 512);
    bn_stats_kernel<<<512, 256, 0, stream>>>(H, stats, (int)T);
    bn_finalize_kernel<<<1, 256, 0, stream>>>(stats, bn1_g, bn1_b, prm, invT);
    bn_apply_kernel<1><<<(TOT + 255) / 256, 256, 0, stream>>>(H, prm, Hb, TOT);

    // qkv GEMM: QKV[t][320]
    gemm_nt_kernel<1><<<dim3(NBLK, 320 / 64), 256, 0, stream>>>(
        Wq[L], Hb, qkv_b, nullptr, QKV, nullptr, 256, 320);

    // attention + residual into H
    attention_kernel<<<NBATCH / 4, 256, 0, stream>>>(QKV, H);

    // bn2 (no relu) -> Hb
    zero_kernel<<<2, 256, 0, stream>>>(stats, 512);
    bn_stats_kernel<<<512, 256, 0, stream>>>(H, stats, (int)T);
    bn_finalize_kernel<<<1, 256, 0, stream>>>(stats, bn2_g, bn2_b, prm, invT);
    bn_apply_kernel<0><<<(TOT + 255) / 256, 256, 0, stream>>>(H, prm, Hb, TOT);

    // d1: Y1 = relu(d1_w @ Hb + d1_b)  (bf16)  [aliases QKV space]
    gemm_nt_kernel<2><<<dim3(NBLK, 256 / 64), 256, 0, stream>>>(
        Wd1[L], Hb, d1_b, nullptr, nullptr, Y1, 256, NCH);
    // d2: H += d2_w @ Y1 + d2_b
    gemm_nt_kernel<3><<<dim3(NBLK, 256 / 64), 256, 0, stream>>>(
        Wd2[L], Y1, d2_b, nullptr, H, nullptr, 256, NCH);
  }

  // ---- head: G1 = relu(f1p @ H + f1_b);  out = 10*(G1 . f2_w + f2_b) ----
  f32_to_bf16_kernel<<<(TOT + 255) / 256, 256, 0, stream>>>(H, Hb, TOT);
  // Hb viewed as [B][1536] (token-major => sample-contiguous)
  gemm_nt_kernel<2><<<dim3(NBATCH / 256, 512 / 64), 256, 0, stream>>>(
      Wf1p, Hb, f1_b, nullptr, nullptr, G1, 1536, 512);
  f2_final_kernel<<<NBATCH / 256, 256, 0, stream>>>(G1, f2_w, f2_b, (float*)d_out);
}
